// BEVMSDeformAttn_45767171506679
// MI455X (gfx1250) — matrix operand order repaired and checked by reading, not compiled, not run on hardware
//
#include <hip/hip_runtime.h>
#include <hip/hip_bf16.h>

#define NQ     6400
#define DMODEL 128
#define NH     8
#define DH     16

typedef __attribute__((ext_vector_type(16))) __bf16       v16bf;
typedef __attribute__((ext_vector_type(8)))  float        v8f;
typedef __attribute__((ext_vector_type(4)))  unsigned int v4u;

union FragBF { v4u q[2]; v16bf b; };          // 32B: one WMMA A/B fragment per lane
union Chunk8 { v4u u; unsigned short s[8]; }; // 16B: 8 bf16

__device__ __forceinline__ unsigned short f2bf(float f) {
  unsigned int u = __float_as_uint(f);
  u += 0x7FFFu + ((u >> 16) & 1u);            // round-to-nearest-even
  return (unsigned short)(u >> 16);
}
__device__ __forceinline__ float bf2f(unsigned short s) {
  return __uint_as_float(((unsigned int)s) << 16);
}
__device__ __forceinline__ int iclamp(int v, int lo, int hi) {
  return v < lo ? lo : (v > hi ? hi : v);
}

// ---------------- elementwise f32 -> bf16 ----------------
__global__ void cvt_bf16_kernel(const float* __restrict__ src,
                                unsigned short* __restrict__ dst, int n) {
  int i = blockIdx.x * 256 + threadIdx.x;
  if (i < n) dst[i] = f2bf(src[i]);
}

// ---------------- CHW f32 -> HWC bf16 ----------------
__global__ void chw2hwc_kernel(const float* __restrict__ src,
                               unsigned short* __restrict__ dst, int C, int HW) {
  int i = blockIdx.x * 256 + threadIdx.x;
  if (i < C * HW) {
    int c = i % C, pix = i / C;
    dst[i] = f2bf(src[c * HW + pix]);
  }
}

// ---------------- dense GEMM: D[M,N] = A[M,K](bf16) @ W[N,K]^T(bf16) + bias ----------------
// one wave per 16x16 output tile, K-loop of v_wmma_f32_16x16x32_bf16
__global__ __launch_bounds__(32) void gemm_bf16_kernel(
    const unsigned short* __restrict__ A, const unsigned short* __restrict__ W,
    const float* __restrict__ bias, float* __restrict__ D, int M, int N, int K) {
  const int lane = threadIdx.x;
  const int lo = lane & 15, hi = lane >> 4;
  const int mrow = blockIdx.x * 16 + lo;   // A fragment row (m = lane%16)
  const int ncol = blockIdx.y * 16 + lo;   // B fragment column / W row (n = lane%16)
  v8f acc = {};
  for (int k0 = 0; k0 < K; k0 += 32) {
    FragBF a, b;
    // A 16x32 bf16 layout: lane(m,hi): elems 0-7 -> K=k0+hi*8.., elems 8-15 -> K=k0+16+hi*8..
    a.q[0] = *(const v4u*)(A + (size_t)mrow * K + k0 + hi * 8);
    a.q[1] = *(const v4u*)(A + (size_t)mrow * K + k0 + 16 + hi * 8);
    // B 32x16 bf16 layout: lane(n,hi): elems 0-7 -> K=k0+hi*8.., elems 8-15 -> K=k0+16+hi*8..
    b.q[0] = *(const v4u*)(W + (size_t)ncol * K + k0 + hi * 8);
    b.q[1] = *(const v4u*)(W + (size_t)ncol * K + k0 + 16 + hi * 8);
    acc = __builtin_amdgcn_wmma_f32_16x16x32_bf16(false, a.b, false, b.b,
                                                  (short)0, acc, false, false);
  }
  const float bv = bias[ncol];
  const int rbase = blockIdx.x * 16 + hi * 8;          // C/D: M = r + 8*(lane/16)
  for (int r = 0; r < 8; ++r)
    D[(size_t)(rbase + r) * N + ncol] = acc[r] + bv;
}

// ---------------- deformable attention core ----------------
// grid = (Lq/16, 8 heads), 32 threads (1 wave). Per level: gather+blend 64 sample
// rows (16q x 4p) of C channels into an LDS A-tile, WMMA against the head's 16xC
// slice of Wk, then logits/softmax/weighted-combine -> tmp (bf16, pre-W_out).
__global__ __launch_bounds__(32) void msda_kernel(
    const float* __restrict__ query, const float* __restrict__ ref,
    const float* __restrict__ off,
    const unsigned short* __restrict__ f0, const unsigned short* __restrict__ f1,
    const unsigned short* __restrict__ f2,
    const unsigned short* __restrict__ Wk0, const unsigned short* __restrict__ Wk1,
    const unsigned short* __restrict__ Wk2,
    const float* __restrict__ bk0, const float* __restrict__ bk1,
    const float* __restrict__ bk2,
    unsigned short* __restrict__ tmp) {
  __shared__ unsigned short sA[64 * 128];  // A-tile: 64 sample rows x C bf16 (16 KB)
  __shared__ float sK[192 * 16];           // projected keys: (l*64+row) x 16  (12 KB)
  __shared__ float sQh[16 * 16];           // this head's 16 query vectors
  __shared__ float sL[16 * 12];            // logits -> attention weights

  const int lane = threadIdx.x;
  const int lo = lane & 15, hi = lane >> 4;
  const int q0 = blockIdx.x * 16;
  const int h  = blockIdx.y;

  for (int i = lane; i < 256; i += 32) {
    int q = i >> 4, d = i & 15;
    sQh[i] = query[(size_t)(q0 + q) * DMODEL + h * DH + d];
  }
  __syncthreads();

  const unsigned short* feats[3] = {f0, f1, f2};
  const unsigned short* Wks[3]   = {Wk0, Wk1, Wk2};
  const float*          bks[3]   = {bk0, bk1, bk2};
  const int HWs[3] = {128, 64, 32};
  const int Cs[3]  = {128, 128, 64};

#pragma unroll
  for (int l = 0; l < 3; ++l) {
    const int Hl = HWs[l], Wl = HWs[l], C = Cs[l];
    const float fH = (float)Hl, fW = (float)Wl;
    const unsigned short* feat = feats[l];

    // --- gather + bilinear blend: 2 rows per lane ---
    for (int rr = 0; rr < 2; ++rr) {
      const int row = lane * 2 + rr;
      const int qloc = row >> 2, p = row & 3;
      const int q = q0 + qloc;
      const size_t ob = (size_t)q * 192 + h * 24 + l * 8 + p * 2;
      float lx = ref[(size_t)q * 6 + l * 2 + 0] + off[ob + 0] / fW;
      float ly = ref[(size_t)q * 6 + l * 2 + 1] + off[ob + 1] / fH;
      float x = lx * fW - 0.5f, y = ly * fH - 0.5f;
      float xf = floorf(x), yf = floorf(y);
      float wx = x - xf, wy = y - yf;
      int x0 = (int)xf, y0 = (int)yf;
      float w[4] = {(1.f - wx) * (1.f - wy), wx * (1.f - wy),
                    (1.f - wx) * wy,         wx * wy};
      const unsigned short* cp[4];
#pragma unroll
      for (int cn = 0; cn < 4; ++cn) {
        int xi = x0 + (cn & 1), yi = y0 + (cn >> 1);
        if (!(xi >= 0 && xi < Wl && yi >= 0 && yi < Hl)) w[cn] = 0.f;
        cp[cn] = feat + (size_t)(iclamp(yi, 0, Hl - 1) * Wl + iclamp(xi, 0, Wl - 1)) * C;
      }
      for (int c = 0; c < C; c += 8) {
        Chunk8 v0, v1, v2, v3;
        v0.u = *(const v4u*)(cp[0] + c);
        v1.u = *(const v4u*)(cp[1] + c);
        v2.u = *(const v4u*)(cp[2] + c);
        v3.u = *(const v4u*)(cp[3] + c);
#pragma unroll
        for (int i = 0; i < 8; ++i) {
          float f = w[0] * bf2f(v0.s[i]) + w[1] * bf2f(v1.s[i]) +
                    w[2] * bf2f(v2.s[i]) + w[3] * bf2f(v3.s[i]);
          sA[row * C + c + i] = f2bf(f);
        }
      }
    }
    __syncthreads();

    // --- WMMA key projection: [64 x C] @ [C x 16] ---
    const int nK = C >> 5;
    FragBF bfr[4];
    const unsigned short* Wk = Wks[l];
    for (int kt = 0; kt < nK; ++kt) {
      // B[k][d] = Wk[h*16+d][k]; lane covers d=lane%16, K runs hi*8.. and 16+hi*8..
      bfr[kt].q[0] = *(const v4u*)(Wk + (size_t)(h * DH + lo) * C + kt * 32 + hi * 8);
      bfr[kt].q[1] = *(const v4u*)(Wk + (size_t)(h * DH + lo) * C + kt * 32 + 16 + hi * 8);
    }
    const float bkv = bks[l][h * DH + lo];
#pragma unroll
    for (int mt = 0; mt < 4; ++mt) {
      v8f acc = {};
      for (int kt = 0; kt < nK; ++kt) {
        FragBF a;
        const int arow = mt * 16 + lo;
        a.q[0] = *(const v4u*)(sA + arow * C + kt * 32 + hi * 8);
        a.q[1] = *(const v4u*)(sA + arow * C + kt * 32 + 16 + hi * 8);
        acc = __builtin_amdgcn_wmma_f32_16x16x32_bf16(false, a.b, false, bfr[kt].b,
                                                      (short)0, acc, false, false);
      }
#pragma unroll
      for (int r = 0; r < 8; ++r) {
        const int row = mt * 16 + r + hi * 8;
        sK[(l * 64 + row) * 16 + lo] = acc[r] + bkv;
      }
    }
    __syncthreads();

    // --- logits: q . k, 2 rows per lane ---
    for (int rr = 0; rr < 2; ++rr) {
      const int row = lane * 2 + rr;
      const int qloc = row >> 2, p = row & 3;
      float a = 0.f;
#pragma unroll
      for (int dd = 0; dd < DH; ++dd)
        a += sQh[qloc * DH + dd] * sK[(l * 64 + row) * 16 + dd];
      sL[qloc * 12 + l * 4 + p] = a;
    }
    __syncthreads();
  }

  // --- softmax over 12 (scale = 1/sqrt(16)) ---
  if (lane < 16) {
    float v[12], m = -1e30f;
#pragma unroll
    for (int j = 0; j < 12; ++j) { v[j] = sL[lane * 12 + j] * 0.25f; m = fmaxf(m, v[j]); }
    float s = 0.f;
#pragma unroll
    for (int j = 0; j < 12; ++j) { v[j] = __expf(v[j] - m); s += v[j]; }
    const float inv = 1.f / s;
#pragma unroll
    for (int j = 0; j < 12; ++j) sL[lane * 12 + j] = v[j] * inv;
  }
  __syncthreads();

  // --- weighted combine -> tmp bf16 (input of the W_out GEMM) ---
  {
    const int q = lo, dbase = hi * 8;
#pragma unroll
    for (int dd = 0; dd < 8; ++dd) {
      const int d = dbase + dd;
      float a = 0.f;
#pragma unroll
      for (int lp = 0; lp < 12; ++lp) {
        const int row = (lp >> 2) * 64 + q * 4 + (lp & 3);
        a += sL[q * 12 + lp] * sK[row * 16 + d];
      }
      tmp[(size_t)(q0 + q) * DMODEL + h * DH + d] = f2bf(a);
    }
  }
}

extern "C" void kernel_launch(void* const* d_in, const int* in_sizes, int n_in,
                              void* d_out, int out_size, void* d_ws, size_t ws_size,
                              hipStream_t stream) {
  const float* query = (const float*)d_in[0];
  const float* refp  = (const float*)d_in[1];
  const float* feat0 = (const float*)d_in[2];
  const float* feat1 = (const float*)d_in[3];
  const float* feat2 = (const float*)d_in[4];
  const float* Woff  = (const float*)d_in[6];
  const float* boff  = (const float*)d_in[7];
  const float* Wk0   = (const float*)d_in[8];
  const float* bk0   = (const float*)d_in[9];
  const float* Wk1   = (const float*)d_in[10];
  const float* bk1   = (const float*)d_in[11];
  const float* Wk2   = (const float*)d_in[12];
  const float* bk2   = (const float*)d_in[13];
  const float* Wout  = (const float*)d_in[14];
  const float* bout  = (const float*)d_in[15];
  float* out = (float*)d_out;

  char* ws = (char*)d_ws;
  size_t o = 0;
  auto alloc = [&](size_t bytes) -> void* {
    o = (o + 255) & ~(size_t)255;
    void* p = ws + o;
    o += bytes;
    return p;
  };
  unsigned short* q_bf   = (unsigned short*)alloc((size_t)NQ * 128 * 2);
  unsigned short* woff_b = (unsigned short*)alloc(192 * 128 * 2);
  unsigned short* wk0_b  = (unsigned short*)alloc(128 * 128 * 2);
  unsigned short* wk1_b  = (unsigned short*)alloc(128 * 128 * 2);
  unsigned short* wk2_b  = (unsigned short*)alloc(128 * 64 * 2);
  unsigned short* wout_b = (unsigned short*)alloc(128 * 128 * 2);
  unsigned short* f0h    = (unsigned short*)alloc((size_t)128 * 128 * 128 * 2);
  unsigned short* f1h    = (unsigned short*)alloc((size_t)64 * 64 * 128 * 2);
  unsigned short* f2h    = (unsigned short*)alloc((size_t)32 * 32 * 64 * 2);
  float*          off_f  = (float*)alloc((size_t)NQ * 192 * 4);
  unsigned short* tmp_bf = (unsigned short*)alloc((size_t)NQ * 128 * 2);

  auto cvt = [&](const float* s, unsigned short* d, int n) {
    cvt_bf16_kernel<<<(n + 255) / 256, 256, 0, stream>>>(s, d, n);
  };
  cvt(query, q_bf, NQ * 128);
  cvt(Woff, woff_b, 192 * 128);
  cvt(Wk0, wk0_b, 128 * 128);
  cvt(Wk1, wk1_b, 128 * 128);
  cvt(Wk2, wk2_b, 128 * 64);
  cvt(Wout, wout_b, 128 * 128);

  chw2hwc_kernel<<<(128 * 16384 + 255) / 256, 256, 0, stream>>>(feat0, f0h, 128, 16384);
  chw2hwc_kernel<<<(128 * 4096 + 255) / 256, 256, 0, stream>>>(feat1, f1h, 128, 4096);
  chw2hwc_kernel<<<(64 * 1024 + 255) / 256, 256, 0, stream>>>(feat2, f2h, 64, 1024);

  // sampling offsets: [6400,192] = query @ W_off^T + b_off
  gemm_bf16_kernel<<<dim3(NQ / 16, 192 / 16), 32, 0, stream>>>(
      q_bf, woff_b, boff, off_f, NQ, 192, 128);

  // gather + key projection + attention -> tmp (bf16)
  msda_kernel<<<dim3(NQ / 16, NH), 32, 0, stream>>>(
      query, refp, off_f, f0h, f1h, f2h, wk0_b, wk1_b, wk2_b, bk0, bk1, bk2, tmp_bf);

  // output projection: [6400,128] = tmp @ W_out^T + b_out
  gemm_bf16_kernel<<<dim3(NQ / 16, 128 / 16), 32, 0, stream>>>(
      tmp_bf, wout_b, bout, out, NQ, 128, 128);
}
